// ARNN_7765300871447
// MI455X (gfx1250) — compile-verified, matrix operand order checked by reading, not run on hardware
//
#include <hip/hip_runtime.h>
#include <hip/hip_bf16.h>

typedef __bf16 bf16_t;
typedef __attribute__((ext_vector_type(16))) __bf16 v16bf;
typedef __attribute__((ext_vector_type(8)))  __bf16 v8bf;
typedef __attribute__((ext_vector_type(8)))  float  v8f;
typedef unsigned int u32x4 __attribute__((ext_vector_type(4)));
typedef int          i32x4 __attribute__((ext_vector_type(4)));
typedef int          i32x8 __attribute__((ext_vector_type(8)));

#define NWG   16
#define TPB   256
#define HDIM  1024
#define BDIM  128
#define TDIM  512
#define IDIM  64
#define ODIM  64

// ---------------- global workspace layout (bytes) ----------------
#define OFF_H    0            // 2 x 128x1024 bf16 (524288)
#define OFF_FGT  524288       // 1024x1024 bf16    (2097152)
#define OFF_REC  2621440      // 1024x1024 bf16    (2097152)
#define OFF_ENC  4718592      // 1024x64 bf16      (131072)
#define OFF_DEC  4849664      // 64x1024 bf16      (131072)
#define OFF_XT   4980736      // 512x128x64 bf16   (8388608)
#define OFF_PART 13369344     // 2 x 16x128x64 f32 (1048576, double-buffered)
#define OFF_SYNC 14417920     // unsigned

// ---------------- dynamic LDS layout (bytes) ----------------
// Big weight slabs are written by TDM with pad_enable: 16B pad every 1024B,
// so a 2048B logical row occupies 2080B (1040 bf16) -> 8-bank row rotation.
#define LDSO_FGT 0            // 64 rows x 2080B = 133120
#define LDSO_REC 133120       // 133120
#define LDSO_ENC 266240       // 64x64 bf16 compact = 8192
#define LDSO_DEC 274432       // 64x64 bf16 compact = 8192
#define LDSO_H   282624       // 128 x 72 bf16 (padded) = 18432
#define SMEM_BYTES 301056

__device__ __forceinline__ v16bf make_frag(v8bf lo, v8bf hi) {
  return __builtin_shufflevector(lo, hi, 0, 1, 2, 3, 4, 5, 6, 7,
                                 8, 9, 10, 11, 12, 13, 14, 15);
}

// plain layout: row pointer + kb, reads [kb..kb+7] and [kb+16..kb+23]
__device__ __forceinline__ v16bf load_frag(const bf16_t* rowp, int kb) {
  v8bf lo = *(const v8bf*)(rowp + kb);
  v8bf hi = *(const v8bf*)(rowp + kb + 16);
  return make_frag(lo, hi);
}

// TDM-padded LDS layout: row stride 1040 elems, +8 elems per 512-elem region
__device__ __forceinline__ v16bf load_frag_pad(const bf16_t* base, int row, int kb) {
  const bf16_t* rowp = base + row * 1040 + ((kb >> 9) << 3);
  v8bf lo = *(const v8bf*)(rowp + kb);
  v8bf hi = *(const v8bf*)(rowp + kb + 16);
  return make_frag(lo, hi);
}

__device__ __forceinline__ v8f wmma_bf16(v16bf a, v16bf b, v8f c) {
  return __builtin_amdgcn_wmma_f32_16x16x32_bf16(false, a, false, b, (short)0, c,
                                                 false, false);
}

// ---- Tensor Data Mover: 2D tile (tile_w x tile_h elems, 2B each) -> LDS ----
__device__ __forceinline__ void tdm_load_2d(unsigned lds_off, const void* gaddr,
                                            unsigned tile_w, unsigned tile_h,
                                            unsigned row_stride, bool pad) {
  const unsigned long long ga = (unsigned long long)(uintptr_t)gaddr;
  u32x4 g0;
  g0[0] = 1u;                                   // count=1, user descriptor
  g0[1] = lds_off;                              // lds_addr (byte offset)
  g0[2] = (unsigned)(ga & 0xFFFFFFFFu);         // global_addr[31:0]
  g0[3] = (unsigned)((ga >> 32) & 0x01FFFFFFu)  // global_addr[56:32]
          | (2u << 30);                         // type = 2 ("image")
  unsigned d0 = (1u << 16);                     // data_size = 1 -> 2 bytes
  if (pad) d0 |= (1u << 20)                     // pad_enable
              |  (7u << 22)                     // pad_interval: 256 DWORDs
              |  (3u << 25);                    // pad_amount: 4 DWORDs (16B)
  i32x8 g1;
  g1[0] = (int)d0;
  g1[1] = (int)((row_stride & 0xFFFFu) << 16);                  // tensor_dim0 lo16 @bit48
  g1[2] = (int)(((row_stride >> 16) & 0xFFFFu) |
                ((tile_h & 0xFFFFu) << 16));                    // dim0 hi | tensor_dim1 lo
  g1[3] = (int)(((tile_h >> 16) & 0xFFFFu) |
                ((tile_w & 0xFFFFu) << 16));                    // dim1 hi | tile_dim0
  g1[4] = (int)(tile_h & 0xFFFFu);                              // tile_dim1 (tile_dim2=0)
  g1[5] = (int)row_stride;                                      // tensor_dim0_stride lo32
  g1[6] = 0;
  g1[7] = 0;
  const i32x4 z4 = {0, 0, 0, 0};
#if __clang_major__ >= 23
  const i32x8 z8 = {0, 0, 0, 0, 0, 0, 0, 0};
  __builtin_amdgcn_tensor_load_to_lds(g0, g1, z4, z4, z8, 0);
#else
  __builtin_amdgcn_tensor_load_to_lds(g0, g1, z4, z4, 0);
#endif
}

__device__ __forceinline__ void grid_barrier(unsigned* cnt, unsigned& no) {
  __syncthreads();
  ++no;
  if (threadIdx.x == 0) {
    __threadfence();
    atomicAdd(cnt, 1u);
    const unsigned target = no * NWG;
    while (*(volatile unsigned*)cnt < target) {
      __builtin_amdgcn_s_sleep(8);
    }
    __threadfence();
  }
  __syncthreads();
}

// ---------------- prep: bf16 weight conversion, h0, sync reset ----------------
__global__ __launch_bounds__(TPB) void prep_misc_kernel(
    const float* __restrict__ fgt_w, const float* __restrict__ rec_w,
    const float* __restrict__ enc_w, const float* __restrict__ dec_w,
    const float* __restrict__ hinit_w, const float* __restrict__ hinit_b,
    bf16_t* __restrict__ fgt_bf, bf16_t* __restrict__ rec_bf,
    bf16_t* __restrict__ enc_bf, bf16_t* __restrict__ dec_bf,
    bf16_t* __restrict__ h_bf, unsigned* __restrict__ sync_cnt) {
  const int i = blockIdx.x * TPB + threadIdx.x;  // grid sized to 1<<20 exactly
  fgt_bf[i] = (bf16_t)fgt_w[i];
  rec_bf[i] = (bf16_t)rec_w[i];
  if (i < HDIM * IDIM) {
    enc_bf[i] = (bf16_t)enc_w[i];
    dec_bf[i] = (bf16_t)dec_w[i];
  }
  if (i < BDIM * HDIM) {
    const int hh = i & (HDIM - 1);
    h_bf[i] = (bf16_t)(hinit_w[hh] + hinit_b[hh]);
  }
  if (i == 0) *sync_cnt = 0u;
}

// x[B,T,I] f32 -> xt[T,B,I] bf16
__global__ __launch_bounds__(TPB) void prep_x_kernel(const float* __restrict__ x,
                                                     bf16_t* __restrict__ xt) {
  const int e = blockIdx.x * TPB + threadIdx.x;  // grid sized to T*B*I exactly
  const int t = e / (BDIM * IDIM);
  const int r = e - t * (BDIM * IDIM);
  const int b = r >> 6;
  const int ii = r & 63;
  xt[e] = (bf16_t)x[(b * TDIM + t) * IDIM + ii];
}

// ---------------- persistent recurrent kernel ----------------
__global__ __launch_bounds__(TPB, 1) void arnn_persistent_kernel(
    bf16_t* __restrict__ h_bf, const bf16_t* __restrict__ fgt_bf,
    const bf16_t* __restrict__ rec_bf, const bf16_t* __restrict__ enc_bf,
    const bf16_t* __restrict__ dec_bf, const bf16_t* __restrict__ xt_bf,
    const float* __restrict__ enc_b, const float* __restrict__ dec_b,
    float* __restrict__ part, unsigned* __restrict__ sync_cnt,
    float* __restrict__ out) {
  const int g = blockIdx.x;         // H-slab owner: columns [g*64, g*64+64)
  const int tid = threadIdx.x;
  const int wv = tid >> 5;          // 8 waves: M rows [16*wv, 16*wv+16)
  const int lane = tid & 31;
  const int lhi = lane >> 4;
  const int llo = lane & 15;
  const int m_row = wv * 16 + llo;
  const int colslab = g * 64;

  extern __shared__ char smem_raw[];
  bf16_t* lds_fgt = (bf16_t*)(smem_raw + LDSO_FGT);
  bf16_t* lds_rec = (bf16_t*)(smem_raw + LDSO_REC);
  bf16_t* lds_enc = (bf16_t*)(smem_raw + LDSO_ENC);
  bf16_t* lds_dec = (bf16_t*)(smem_raw + LDSO_DEC);
  bf16_t* lds_h   = (bf16_t*)(smem_raw + LDSO_H);

  // ---- TDM prefill: park this WG's weight slabs in LDS for all 512 steps ----
  if (wv == 0) {
    tdm_load_2d((unsigned)(uintptr_t)lds_fgt, fgt_bf + (size_t)colslab * HDIM,
                HDIM, 64, HDIM, true);
  } else if (wv == 1) {
    tdm_load_2d((unsigned)(uintptr_t)lds_rec, rec_bf + (size_t)colslab * HDIM,
                HDIM, 64, HDIM, true);
  } else if (wv == 2) {
    tdm_load_2d((unsigned)(uintptr_t)lds_enc, enc_bf + (size_t)colslab * IDIM,
                IDIM, 64, IDIM, false);
  } else if (wv == 3) {
    tdm_load_2d((unsigned)(uintptr_t)lds_dec, dec_bf + colslab,
                64, 64, HDIM, false);
  }
  // preload h0 slab into LDS staging (gate update reads h_old from LDS)
  for (int e = tid; e < BDIM * 64; e += TPB) {
    const int row = e >> 6;
    const int c = e & 63;
    lds_h[row * 72 + c] = h_bf[row * HDIM + colslab + c];
  }
  __builtin_amdgcn_s_wait_tensorcnt(0);
  __syncthreads();

  unsigned sync_no = 0;

  for (int t = 0; t < TDIM; ++t) {
    const bf16_t* hcur = h_bf + (size_t)(t & 1) * (BDIM * HDIM);
    bf16_t* hnext = h_bf + (size_t)((t + 1) & 1) * (BDIM * HDIM);
    const bf16_t* xt_t = xt_bf + (size_t)t * (BDIM * IDIM);
    float* part_t = part + (size_t)(t & 1) * (NWG * BDIM * ODIM);

    if ((t + 1) < TDIM && tid == 0)
      __builtin_prefetch(xt_bf + (size_t)(t + 1) * (BDIM * IDIM), 0, 1);

    // ---- encoder slab: x_t(128x64) @ enc_w_slab^T, K=64 (B from LDS) ----
    v8f eacc[4];
    {
      const int kb = lhi * 8;
      const v16bf a = load_frag(xt_t + m_row * IDIM, kb);
#pragma unroll
      for (int nt = 0; nt < 4; ++nt)
        eacc[nt] = wmma_bf16(a, load_frag(lds_enc + (nt * 16 + llo) * IDIM, kb),
                             v8f{});
    }
    {
      const int kb = 32 + lhi * 8;
      const v16bf a = load_frag(xt_t + m_row * IDIM, kb);
#pragma unroll
      for (int nt = 0; nt < 4; ++nt)
        eacc[nt] = wmma_bf16(a, load_frag(lds_enc + (nt * 16 + llo) * IDIM, kb),
                             eacc[nt]);
    }

    // ---- recurrent GEMMs: A = h (global, L2-hot), B = LDS slabs, K=1024 ----
    v8f facc[4], racc[4];
    {  // peeled kc=0: C = inline 0 (no zero-init mov storm)
      const int kb = lhi * 8;
      const v16bf a = load_frag(hcur + m_row * HDIM, kb);
#pragma unroll
      for (int nt = 0; nt < 4; ++nt) {
        const int brow = nt * 16 + llo;
        facc[nt] = wmma_bf16(a, load_frag_pad(lds_fgt, brow, kb), v8f{});
        racc[nt] = wmma_bf16(a, load_frag_pad(lds_rec, brow, kb), v8f{});
      }
    }
#pragma unroll 2
    for (int kc = 1; kc < 32; ++kc) {
      const int kb = kc * 32 + lhi * 8;
      const v16bf a = load_frag(hcur + m_row * HDIM, kb);
#pragma unroll
      for (int nt = 0; nt < 4; ++nt) {
        const int brow = nt * 16 + llo;
        facc[nt] = wmma_bf16(a, load_frag_pad(lds_fgt, brow, kb), facc[nt]);
        racc[nt] = wmma_bf16(a, load_frag_pad(lds_rec, brow, kb), racc[nt]);
      }
    }

    // ---- gates + hidden update; h_old from LDS; write global bf16 + LDS ----
#pragma unroll
    for (int nt = 0; nt < 4; ++nt) {
      const int col_l = nt * 16 + llo;
      const int colg = colslab + col_l;
      const float eb = enc_b[colg];
#pragma unroll
      for (int j = 0; j < 8; ++j) {
        const int row = wv * 16 + j + lhi * 8;  // C/D layout: VGPR j -> M=j (+8 hi lanes)
        const float hold = (float)lds_h[row * 72 + col_l];  // own element, no race
        const float f = 1.0f / (1.0f + __expf(-facc[nt][j]));
        const float pre = eacc[nt][j] + eb + racc[nt][j];
        const float hn = pre / (1.0f + fabsf(pre));
        const float hv = (1.0f - f) * hold + f * hn;
        const bf16_t hb = (bf16_t)hv;
        hnext[row * HDIM + colg] = hb;
        lds_h[row * 72 + col_l] = hb;
      }
    }
    __syncthreads();

    // ---- decoder partial: h_slab(128x64) @ dec_w[:, slab]^T, K=64 ----
    v8f dacc[4];
    {
      const int kb = lhi * 8;
      const v16bf a = load_frag(lds_h + m_row * 72, kb);
#pragma unroll
      for (int nt = 0; nt < 4; ++nt)
        dacc[nt] = wmma_bf16(a, load_frag(lds_dec + (nt * 16 + llo) * 64, kb),
                             v8f{});
    }
    {
      const int kb = 32 + lhi * 8;
      const v16bf a = load_frag(lds_h + m_row * 72, kb);
#pragma unroll
      for (int nt = 0; nt < 4; ++nt)
        dacc[nt] = wmma_bf16(a, load_frag(lds_dec + (nt * 16 + llo) * 64, kb),
                             dacc[nt]);
    }
    float* partg = part_t + g * (BDIM * ODIM);
#pragma unroll
    for (int nt = 0; nt < 4; ++nt) {
#pragma unroll
      for (int j = 0; j < 8; ++j) {
        const int row = wv * 16 + j + lhi * 8;
        partg[row * ODIM + nt * 16 + llo] = dacc[nt][j];
      }
    }

    // single barrier per step: part is double-buffered, h is 2-deep ping-pong
    grid_barrier(sync_cnt, sync_no);

    // ---- cross-WG reduction of decoder partials -> out[t] ----
    for (int e = g * TPB + tid; e < BDIM * ODIM; e += NWG * TPB) {
      float s = dec_b[e & 63];
#pragma unroll
      for (int gg = 0; gg < NWG; ++gg) s += part_t[gg * (BDIM * ODIM) + e];
      out[(size_t)t * (BDIM * ODIM) + e] = s;
    }
  }

  // ---- final hidden (f32) appended after output[T,B,O] ----
  const bf16_t* hfin = h_bf + (size_t)(TDIM & 1) * (BDIM * HDIM);
  float* hout = out + (size_t)TDIM * (BDIM * ODIM);
  for (int e = tid; e < BDIM * 64; e += TPB) {
    const int row = e >> 6;
    const int c = colslab + (e & 63);
    hout[row * HDIM + c] = (float)hfin[row * HDIM + c];
  }
}

extern "C" void kernel_launch(void* const* d_in, const int* in_sizes, int n_in,
                              void* d_out, int out_size, void* d_ws, size_t ws_size,
                              hipStream_t stream) {
  const float* x       = (const float*)d_in[0];
  const float* enc_w   = (const float*)d_in[1];
  const float* enc_b   = (const float*)d_in[2];
  const float* rec_w   = (const float*)d_in[3];
  const float* fgt_w   = (const float*)d_in[4];
  const float* dec_w   = (const float*)d_in[5];
  const float* dec_b   = (const float*)d_in[6];
  const float* hinit_w = (const float*)d_in[7];
  const float* hinit_b = (const float*)d_in[8];
  float* out = (float*)d_out;

  char* ws = (char*)d_ws;
  bf16_t*   h_bf   = (bf16_t*)(ws + OFF_H);
  bf16_t*   fgt_bf = (bf16_t*)(ws + OFF_FGT);
  bf16_t*   rec_bf = (bf16_t*)(ws + OFF_REC);
  bf16_t*   enc_bf = (bf16_t*)(ws + OFF_ENC);
  bf16_t*   dec_bf = (bf16_t*)(ws + OFF_DEC);
  bf16_t*   xt_bf  = (bf16_t*)(ws + OFF_XT);
  float*    part   = (float*)(ws + OFF_PART);
  unsigned* sync   = (unsigned*)(ws + OFF_SYNC);

  prep_misc_kernel<<<(HDIM * HDIM) / TPB, TPB, 0, stream>>>(
      fgt_w, rec_w, enc_w, dec_w, hinit_w, hinit_b,
      fgt_bf, rec_bf, enc_bf, dec_bf, h_bf, sync);

  prep_x_kernel<<<(TDIM * BDIM * IDIM) / TPB, TPB, 0, stream>>>(x, xt_bf);

  arnn_persistent_kernel<<<NWG, TPB, SMEM_BYTES, stream>>>(
      h_bf, fgt_bf, rec_bf, enc_bf, dec_bf, xt_bf, enc_b, dec_b, part, sync, out);
}
